// LocalEnergyCore_16561393894086
// MI455X (gfx1250) — matrix operand order repaired and checked by reading, not compiled, not run on hardware
//
#include <hip/hip_runtime.h>

typedef __attribute__((ext_vector_type(16))) _Float16 v16h;
typedef __attribute__((ext_vector_type(8)))  float    v8f;

#define ZB     32      // batches
#define ZK     8       // channels
#define ZH     512     // height (pow2)
#define ZW     512     // width  (pow2)
#define NS     50      // sites
#define CTXLEN 71      // 9*ZK - 1
#define KPAD   96      // K padded to 3 x 32 for wmma 16x16x32
#define HID    64
#define NSAMP  (ZB * NS)   // 1600
#define MTILE  16
#define NKT    3       // K-steps of 32
#define NNT    4       // N-tiles of 16

__global__ void le_init(unsigned int* c) {
    if (threadIdx.x == 0) *c = 0u;
}

__global__ void le_fini(const unsigned int* __restrict__ c, float* __restrict__ out) {
    if (threadIdx.x == 0) out[0] = (float)(*c) * (1.0f / (float)NSAMP);
}

__global__ __launch_bounds__(32) void le_main(
    const float* __restrict__ z,   const float* __restrict__ W1,
    const float* __restrict__ b1,  const float* __restrict__ W2,
    const float* __restrict__ b2,  const int* __restrict__ b_idx,
    const int* __restrict__ i_idx, const int* __restrict__ j_idx,
    unsigned int* __restrict__ errCount)
{
    // Operands staged directly in WMMA per-lane register order:
    // each lane's 16 f16 live in one contiguous 32B row -> 2x ds_load_b128.
    __shared__ __attribute__((aligned(32))) _Float16 aPack[NKT][32][16];
    __shared__ __attribute__((aligned(32))) _Float16 bPack[NNT][NKT][32][16];
    __shared__ float        actualS[MTILE];
    __shared__ float        partLDS[32][8];
    __shared__ unsigned int errS[MTILE];

    const int lane     = threadIdx.x;
    const int tileBase = blockIdx.x * MTILE;

    // ---- stage W1 (f32 -> f16) scattered into B WMMA layout, zero-pad K ----
    for (int idx = lane; idx < KPAD * HID; idx += 32) {
        const int k = idx / HID, n = idx % HID;
        const _Float16 val = (k < CTXLEN) ? (_Float16)W1[k * HID + n] : (_Float16)0.f;
        // inverse of B 32x16 layout: kl = hi*16 + 2v + e
        const int nt = n >> 4, nc = n & 15;
        const int kk = k >> 5, kl = k & 31;
        const int hi = kl >> 4, r = kl & 15;
        const int v = r >> 1, e = r & 1;
        bPack[nt][kk][hi * 16 + nc][2 * v + e] = val;
    }

    // ---- gather A tile scattered into A WMMA layout (center removed) ----
    for (int idx = lane; idx < MTILE * KPAD; idx += 32) {
        const int m = idx / KPAD, t = idx % KPAD;
        _Float16 val = (_Float16)0.f;
        const int flat = tileBase + m;
        if (flat < NSAMP && t < CTXLEN) {
            const int b  = flat / NS, s = flat - b * NS;
            const int bi = b_idx[s], i0 = i_idx[s], j0 = j_idx[s];
            const int rm  = 4 * ZK + bi;                 // removed center index
            const int src = t + (t >= rm ? 1 : 0);       // skip over rm
            const int n9 = src >> 3, k = src & 7;        // neighbor, channel
            const int di = n9 / 3 - 1, dj = n9 % 3 - 1;
            const int ii = (i0 + di + ZH) & (ZH - 1);
            const int jj = (j0 + dj + ZW) & (ZW - 1);
            val = (_Float16)z[(((size_t)b * ZK + k) * ZH + ii) * ZW + jj];
        }
        // contiguous k order, as B: lane = group*16 + m, element = w = kl & 15
        const int kk = t >> 5, kl = t & 31;
        const int group = kl >> 4, w = kl & 15;
        const int hi = w >> 3, rem = w & 7;
        const int v = group * 4 + (rem >> 1), e = rem & 1;
        aPack[kk][group * 16 + m][w] = val;
    }

    if (lane < MTILE) {
        const int flat = tileBase + lane;
        float a = 0.f;
        if (flat < NSAMP) {
            const int b = flat / NS, s = flat - b * NS;
            a = z[(((size_t)b * ZK + b_idx[s]) * ZH + i_idx[s]) * ZW + j_idx[s]];
        }
        actualS[lane] = a;
    }
    __syncthreads();

    // ---- WMMA GEMM: (16 x 96) @ (96 x 64), f16 in / f32 acc ----
    // Uniform control flow, all 32 lanes active (EXEC all-ones).
    const int ncolLo = lane & 15;

    float part[8];
#pragma unroll
    for (int r = 0; r < 8; ++r) part[r] = 0.f;

#pragma unroll
    for (int nt = 0; nt < NNT; ++nt) {
        v8f acc = {};
#pragma unroll
        for (int kk = 0; kk < NKT; ++kk) {
            const v16h a = *(const v16h*)(&aPack[kk][lane][0]);
            const v16h b = *(const v16h*)(&bPack[nt][kk][lane][0]);
            acc = __builtin_amdgcn_wmma_f32_16x16x32_f16(
                /*neg_a=*/false, a, /*neg_b=*/false, b,
                /*c_mod=*/(short)0, acc, /*reuse_a=*/false, /*reuse_b=*/false);
        }
        // C layout: lane -> N = ncol, VGPR r -> M = r + 8*(lane>>4).
        const int   ncol = nt * 16 + ncolLo;
        const float w2v  = W2[ncol];
        const float b1v  = b1[ncol];
#pragma unroll
        for (int r = 0; r < 8; ++r) {
            float hv = acc[r] + b1v;
            hv = hv > 0.f ? hv : 0.f;
            part[r] += hv * w2v;
        }
    }

#pragma unroll
    for (int r = 0; r < 8; ++r) partLDS[lane][r] = part[r];
    __syncthreads();

    // ---- logits, prediction, integer error count (deterministic) ----
    if (lane < MTILE) {
        const int mm   = lane;
        const int r    = mm & 7;
        const int base = (mm < 8) ? 0 : 16;   // lanes whose VGPR r maps to this M
        float sum = b2[0];
#pragma unroll
        for (int l = 0; l < 16; ++l) sum += partLDS[base + l][r];
        const float pred = (sum > 0.f) ? 1.f : 0.f;
        errS[mm] = (pred != actualS[mm]) ? 1u : 0u;
    }
    __syncthreads();

    if (lane == 0) {
        unsigned int tot = 0;
#pragma unroll
        for (int mm = 0; mm < MTILE; ++mm) tot += errS[mm];
        atomicAdd(errCount, tot);
    }
}

extern "C" void kernel_launch(void* const* d_in, const int* in_sizes, int n_in,
                              void* d_out, int out_size, void* d_ws, size_t ws_size,
                              hipStream_t stream) {
    (void)in_sizes; (void)n_in; (void)out_size; (void)ws_size;
    const float* z     = (const float*)d_in[0];
    const float* W1    = (const float*)d_in[1];
    const float* b1    = (const float*)d_in[2];
    const float* W2    = (const float*)d_in[3];
    const float* b2    = (const float*)d_in[4];
    const int*   b_idx = (const int*)d_in[5];
    const int*   i_idx = (const int*)d_in[6];
    const int*   j_idx = (const int*)d_in[7];

    unsigned int* errCount = (unsigned int*)d_ws;
    float*        out      = (float*)d_out;

    le_init<<<1, 1, 0, stream>>>(errCount);
    le_main<<<NSAMP / MTILE, 32, 0, stream>>>(z, W1, b1, W2, b2,
                                              b_idx, i_idx, j_idx, errCount);
    le_fini<<<1, 1, 0, stream>>>(errCount, out);
}
